// MMDLoss_13761075216961
// MI455X (gfx1250) — matrix operand order repaired and checked by reading, not compiled, not run on hardware
//
#include <hip/hip_runtime.h>
#include <hip/hip_bf16.h>
#include <stdint.h>

// ---------------------------------------------------------------------------
// MMD loss, MI455X (gfx1250). X = concat(source,target): 8192 x 256 f32.
// result = (1/(5*4096^2)) * sum_ij s_i*s_j * sum_b exp(coef_b * L2_ij)
// L2_ij = sq_i + sq_j - 2*(X X^T)_ij ; Gram via split-bf16 WMMA (hi/lo, 3 MACs).
// Round 3: tile staging via Tensor Data Mover (tensor_load_to_lds) with the
// descriptor pad feature reproducing the padded LDS layout (16+4 dwords/row).
// Register-prefetch pipeline removed (it spilled to scratch in round 2).
// ---------------------------------------------------------------------------

typedef __bf16 v16bf __attribute__((ext_vector_type(16)));
typedef float  v8f   __attribute__((ext_vector_type(8)));
typedef unsigned u32x4 __attribute__((ext_vector_type(4)));
typedef int      i32x4 __attribute__((ext_vector_type(4)));
typedef int      i32x8 __attribute__((ext_vector_type(8)));

#define NROWS 8192
#define NS    4096
#define DDIM  256
#define TILE  128
#define KSTEP 32
#define NCHUNK (DDIM / KSTEP)   // 8 k-chunks
#define LROW  20                // u32 stride per LDS tile row: 16 data + 4 pad
#define PLANE_U32 (TILE * 16)   // 2048 dwords per (tile x k-chunk) plane

#if defined(__gfx1250__) && __has_builtin(__builtin_amdgcn_tensor_load_to_lds)
#define USE_TDM 1
#else
#define USE_TDM 0
#endif

__device__ __forceinline__ const float* row_ptr(const float* s, const float* t, int i) {
  return (i < NS) ? (s + (size_t)i * DDIM) : (t + (size_t)(i - NS) * DDIM);
}

__device__ __forceinline__ unsigned f32_to_bf16_rne(float f) {
  unsigned u = __float_as_uint(f);
  unsigned r = 0x7FFFu + ((u >> 16) & 1u);
  return (u + r) >> 16;                 // low 16 bits valid
}
__device__ __forceinline__ float bf16_to_f32(unsigned h) {
  return __uint_as_float(h << 16);
}

// ---------------- kernel 1: row stats + one-time hi/lo split ---------------
__global__ void k_rowstats(const float* __restrict__ src, const float* __restrict__ tgt,
                           float* __restrict__ sq, float* __restrict__ colblk,
                           float* __restrict__ sqblk,
                           unsigned* __restrict__ hiX, unsigned* __restrict__ loX) {
  __shared__ float rows[8][256];
  __shared__ float wsum[8];
  const int tid = threadIdx.x, w = tid >> 5, lane = tid & 31;
  const int i = blockIdx.x * 8 + w;
  const float* p = row_ptr(src, tgt, i) + lane * 8;
  float4 a = *reinterpret_cast<const float4*>(p);
  float4 b = *reinterpret_cast<const float4*>(p + 4);
  rows[w][lane * 8 + 0] = a.x; rows[w][lane * 8 + 1] = a.y;
  rows[w][lane * 8 + 2] = a.z; rows[w][lane * 8 + 3] = a.w;
  rows[w][lane * 8 + 4] = b.x; rows[w][lane * 8 + 5] = b.y;
  rows[w][lane * 8 + 6] = b.z; rows[w][lane * 8 + 7] = b.w;

  float v[8] = {a.x, a.y, a.z, a.w, b.x, b.y, b.z, b.w};
  unsigned hu[8], lu[8];
  #pragma unroll
  for (int e = 0; e < 8; ++e) {
    hu[e] = f32_to_bf16_rne(v[e]);
    lu[e] = f32_to_bf16_rne(v[e] - bf16_to_f32(hu[e]));
  }
  const int chunk = lane >> 2;                 // (lane*8)/32
  const int u     = (lane & 3) * 4;
  const int base  = (chunk * NROWS + i) * 16 + u;
  *reinterpret_cast<uint4*>(hiX + base) =
      make_uint4(hu[0] | (hu[1] << 16), hu[2] | (hu[3] << 16),
                 hu[4] | (hu[5] << 16), hu[6] | (hu[7] << 16));
  *reinterpret_cast<uint4*>(loX + base) =
      make_uint4(lu[0] | (lu[1] << 16), lu[2] | (lu[3] << 16),
                 lu[4] | (lu[5] << 16), lu[6] | (lu[7] << 16));

  float s = a.x*a.x + a.y*a.y + a.z*a.z + a.w*a.w
          + b.x*b.x + b.y*b.y + b.z*b.z + b.w*b.w;
  #pragma unroll
  for (int off = 16; off > 0; off >>= 1) s += __shfl_xor(s, off);
  if (lane == 0) { sq[i] = s; wsum[w] = s; }
  __syncthreads();
  float cp = 0.f;
  #pragma unroll
  for (int r = 0; r < 8; ++r) cp += rows[r][tid];
  colblk[blockIdx.x * 256 + tid] = cp;
  if (tid == 0) {
    float t = 0.f;
    #pragma unroll
    for (int r = 0; r < 8; ++r) t += wsum[r];
    sqblk[blockIdx.x] = t;
  }
}

// --------------------- kernel 2: bandwidth -> coefficients -----------------
__global__ void k_bwcoef(const float* __restrict__ colblk, const float* __restrict__ sqblk,
                         float* __restrict__ coef) {
  __shared__ float red[256];
  const int t = threadIdx.x;
  float cs = 0.f;
  for (int b = 0; b < 1024; ++b) cs += colblk[b * 256 + t];
  red[t] = cs * cs;
  __syncthreads();
  for (int s = 128; s > 0; s >>= 1) { if (t < s) red[t] += red[t + s]; __syncthreads(); }
  const float csq = red[0];
  __syncthreads();
  red[t] = sqblk[t*4] + sqblk[t*4+1] + sqblk[t*4+2] + sqblk[t*4+3];
  __syncthreads();
  for (int s = 128; s > 0; s >>= 1) { if (t < s) red[t] += red[t + s]; __syncthreads(); }
  if (t == 0) {
    const float sumSq = red[0];
    const float n = (float)NROWS;
    const float sumL2 = 2.f * n * sumSq - 2.f * csq;
    const float bw = sqrtf(sumL2 / (n * n - n));
    float mul = 0.25f;
    #pragma unroll
    for (int b = 0; b < 5; ++b) {
      const float bwb = bw * mul;
      coef[b] = -1.f / (2.f * bwb * bwb);
      mul *= 2.f;
    }
  }
}

// ------------------------- main WMMA kernel --------------------------------
#if USE_TDM
// One TDM descriptor copies a contiguous 2048-dword plane into LDS, inserting
// 4 pad dwords after every 16 data dwords -> LDS row stride LROW=20.
__device__ __forceinline__ void tdm_load_plane(const unsigned* gsrc, unsigned ldsByteOff) {
  const unsigned long long ga = (unsigned long long)(const void*)gsrc;
  u32x4 g0;
  g0[0] = 1u;                                  // count=1, user descriptor
  g0[1] = ldsByteOff;                          // D#.lds_addr
  g0[2] = (unsigned)ga;                        // global_addr[31:0]
  g0[3] = (unsigned)(ga >> 32) | (2u << 30);   // global_addr[56:32] | type=2
  i32x8 g1;
  g1[0] = (int)((2u << 16)       // data_size = 4 bytes
              | (1u << 20)       // pad_enable
              | (3u << 22)       // pad_interval: 16 dwords
              | (3u << 25));     // pad_amount : 4 dwords
  g1[1] = (int)((unsigned)PLANE_U32 << 16);    // tensor_dim0 = 2048 (lo16 here)
  g1[2] = (int)(1u << 16);                     // tensor_dim0 hi=0 | tensor_dim1=1
  g1[3] = (int)((unsigned)PLANE_U32 << 16);    // tile_dim0 = 2048
  g1[4] = 1;                                   // tile_dim1 = 1
  g1[5] = PLANE_U32;                           // tensor_dim0_stride = 2048
  g1[6] = 0; g1[7] = 0;
  const i32x4 z4 = {0, 0, 0, 0};
#if __clang_major__ >= 23
  const i32x8 z8 = {0, 0, 0, 0, 0, 0, 0, 0};
  __builtin_amdgcn_tensor_load_to_lds(g0, g1, z4, z4, z8, 0);
#else
  __builtin_amdgcn_tensor_load_to_lds(g0, g1, z4, z4, 0);
#endif
}
__device__ __forceinline__ void tdm_wait() {
#if __has_builtin(__builtin_amdgcn_s_wait_tensorcnt)
  __builtin_amdgcn_s_wait_tensorcnt(0);
#else
  asm volatile("s_wait_tensorcnt 0x0" ::: "memory");
#endif
}
#else
// Fallback: synchronous b128 copy global -> LDS (no register pipelining).
__device__ __forceinline__ void fill_plane(const unsigned* __restrict__ g,
                                           unsigned* s, int tid) {
  const int r0 = tid >> 2, cc = (tid & 3) * 4;
  uint4 a = *reinterpret_cast<const uint4*>(g + tid * 4);
  uint4 b = *reinterpret_cast<const uint4*>(g + (tid + 256) * 4);
  *reinterpret_cast<uint4*>(s + r0 * LROW + cc) = a;
  *reinterpret_cast<uint4*>(s + (r0 + 64) * LROW + cc) = b;
}
#endif

// A frag (16x32, 16-bit): lane m=l%16, half h=l/16; VGPR r: K = 16*(r/4)+8*h+2*(r%4)+{0,1}
__device__ __forceinline__ v16bf load_fragA(const unsigned* base, int row, int h) {
  union { uint4 q[2]; v16bf v; } f;
  f.q[0] = *reinterpret_cast<const uint4*>(base + row * LROW + h * 4);
  f.q[1] = *reinterpret_cast<const uint4*>(base + row * LROW + 8 + h * 4);
  return f.v;
}
// B frag (32x16): lane n=l%16, half h=l/16; VGPR r: K = 16*(r/4)+8*h+2*(r%4)+{0,1}
__device__ __forceinline__ v16bf load_fragB(const unsigned* base, int row, int h) {
  union { uint4 q[2]; v16bf v; } f;
  f.q[0] = *reinterpret_cast<const uint4*>(base + row * LROW + h * 4);
  f.q[1] = *reinterpret_cast<const uint4*>(base + row * LROW + 8 + h * 4);
  return f.v;
}

__global__ void __launch_bounds__(256)
k_mmd(const unsigned* __restrict__ hiX, const unsigned* __restrict__ loX,
      const float* __restrict__ sq, const float* __restrict__ coef,
      float* __restrict__ blockSums) {
  __shared__ unsigned sHiA[TILE * LROW], sLoA[TILE * LROW];
  __shared__ unsigned sHiB[TILE * LROW], sLoB[TILE * LROW];
  __shared__ float red[256];

  const int tid  = threadIdx.x;
  const int w    = tid >> 5, lane = tid & 31;
  const int mg   = w & 3;          // wave's 32-row group within 128
  const int ng   = w >> 2;         // wave's 64-col group within 128
  const int h    = lane >> 4;      // half-wave
  const int l16  = lane & 15;
  const int ibase = blockIdx.y * TILE;
  const int jbase = blockIdx.x * TILE;
  const float sign = ((ibase < NS) == (jbase < NS)) ? 1.f : -1.f;
  const int chunkStride = NROWS * 16;          // u32 per k-chunk plane set

  v8f acc[2][4];
  #pragma unroll
  for (int mt = 0; mt < 2; ++mt)
    #pragma unroll
    for (int nt = 0; nt < 4; ++nt) acc[mt][nt] = (v8f){};

  int aBase = ibase * 16;                      // chunk 0
  int bBase = jbase * 16;

#if USE_TDM
  const unsigned ldsHiA = (unsigned)(size_t)(void*)sHiA;
  const unsigned ldsLoA = (unsigned)(size_t)(void*)sLoA;
  const unsigned ldsHiB = (unsigned)(size_t)(void*)sHiB;
  const unsigned ldsLoB = (unsigned)(size_t)(void*)sLoB;
  if (w == 0) {
    tdm_load_plane(hiX + aBase, ldsHiA);
    tdm_load_plane(loX + aBase, ldsLoA);
    tdm_load_plane(hiX + bBase, ldsHiB);
    tdm_load_plane(loX + bBase, ldsLoB);
    tdm_wait();
  }
#else
  fill_plane(hiX + aBase, sHiA, tid);
  fill_plane(loX + aBase, sLoA, tid);
  fill_plane(hiX + bBase, sHiB, tid);
  fill_plane(loX + bBase, sLoB, tid);
#endif
  __syncthreads();

  for (int c = 0; c < NCHUNK; ++c) {
    v16bf aH[2], aL[2];
    #pragma unroll
    for (int mt = 0; mt < 2; ++mt) {
      const int row = mg * 32 + mt * 16 + l16;
      aH[mt] = load_fragA(sHiA, row, h);
      aL[mt] = load_fragA(sLoA, row, h);
    }
    #pragma unroll
    for (int nt = 0; nt < 4; ++nt) {
      const int col = ng * 64 + nt * 16 + l16;
      const v16bf bH = load_fragB(sHiB, col, h);
      const v16bf bL = load_fragB(sLoB, col, h);
      #pragma unroll
      for (int mt = 0; mt < 2; ++mt) {
        acc[mt][nt] = __builtin_amdgcn_wmma_f32_16x16x32_bf16(
            false, aH[mt], false, bH, (short)0, acc[mt][nt], false, false);
        acc[mt][nt] = __builtin_amdgcn_wmma_f32_16x16x32_bf16(
            false, aH[mt], false, bL, (short)0, acc[mt][nt], false, false);
        acc[mt][nt] = __builtin_amdgcn_wmma_f32_16x16x32_bf16(
            false, aL[mt], false, bH, (short)0, acc[mt][nt], false, false);
      }
    }
    __syncthreads();                           // all waves done reading LDS
    if (c + 1 < NCHUNK) {
      aBase += chunkStride; bBase += chunkStride;
#if USE_TDM
      if (w == 0) {
        tdm_load_plane(hiX + aBase, ldsHiA);
        tdm_load_plane(loX + aBase, ldsLoA);
        tdm_load_plane(hiX + bBase, ldsHiB);
        tdm_load_plane(loX + bBase, ldsLoB);
        tdm_wait();
      }
#else
      fill_plane(hiX + aBase, sHiA, tid);
      fill_plane(loX + aBase, sLoA, tid);
      fill_plane(hiX + bBase, sHiB, tid);
      fill_plane(loX + bBase, sLoB, tid);
#endif
      __syncthreads();                         // new tiles visible
    }
  }

  // Fused epilogue: L2 -> 5 Gaussians -> signed partial sum.
  float c0 = coef[0], c1 = coef[1], c2 = coef[2], c3 = coef[3], c4 = coef[4];
  float part = 0.f;
  #pragma unroll
  for (int mt = 0; mt < 2; ++mt) {
    const int i0 = ibase + mg * 32 + mt * 16 + 8 * h;   // D: M = vgpr + 8*h
    float sqi[8];
    #pragma unroll
    for (int e = 0; e < 8; ++e) sqi[e] = sq[i0 + e];
    #pragma unroll
    for (int nt = 0; nt < 4; ++nt) {
      const int j = jbase + ng * 64 + nt * 16 + l16;    // D: N = lane%16
      const float sqj = sq[j];
      #pragma unroll
      for (int e = 0; e < 8; ++e) {
        float l2 = sqi[e] + sqj - 2.f * acc[mt][nt][e];
        l2 = fmaxf(l2, 0.f);
        part += __expf(c0 * l2) + __expf(c1 * l2) + __expf(c2 * l2)
              + __expf(c3 * l2) + __expf(c4 * l2);
      }
    }
  }

  red[tid] = part;
  __syncthreads();
  for (int s = 128; s > 0; s >>= 1) { if (tid < s) red[tid] += red[tid + s]; __syncthreads(); }
  if (tid == 0) blockSums[blockIdx.y * (NROWS / TILE) + blockIdx.x] = red[0] * sign;
}

// ------------------------- kernel 4: fixed-order finish --------------------
__global__ void k_finish(const float* __restrict__ blockSums, float* __restrict__ out) {
  __shared__ float red[256];
  const int t = threadIdx.x;
  float s = 0.f;
  #pragma unroll
  for (int q = 0; q < 16; ++q) s += blockSums[q * 256 + t];
  red[t] = s;
  __syncthreads();
  for (int k = 128; k > 0; k >>= 1) { if (t < k) red[t] += red[t + k]; __syncthreads(); }
  if (t == 0) out[0] = red[0] * (1.f / (5.f * (float)NS * (float)NS));
}

// ---------------------------------------------------------------------------
extern "C" void kernel_launch(void* const* d_in, const int* in_sizes, int n_in,
                              void* d_out, int out_size, void* d_ws, size_t ws_size,
                              hipStream_t stream) {
  const float* src = (const float*)d_in[0];
  const float* tgt = (const float*)d_in[1];
  float* out = (float*)d_out;

  float* w      = (float*)d_ws;
  float* sq     = w;                     // 8192
  float* colblk = sq + NROWS;            // 1024 * 256
  float* sqblk  = colblk + 1024 * 256;   // 1024
  float* coef   = sqblk + 1024;          // 5 (+3 pad)
  float* bsums  = coef + 8;              // 4096
  unsigned* hiX = (unsigned*)(bsums + NROWS / TILE * (NROWS / TILE));
  unsigned* loX = hiX + NCHUNK * NROWS * 16;   // 4 MB each

  k_rowstats<<<NROWS / 8, 256, 0, stream>>>(src, tgt, sq, colblk, sqblk, hiX, loX);
  k_bwcoef<<<1, 256, 0, stream>>>(colblk, sqblk, coef);
  dim3 grid(NROWS / TILE, NROWS / TILE);
  k_mmd<<<grid, 256, 0, stream>>>(hiX, loX, sq, coef, bsums);
  k_finish<<<1, 256, 0, stream>>>(bsums, out);
}
